// Attention_670014898511
// MI455X (gfx1250) — matrix operand order repaired and checked by reading, not compiled, not run on hardware
//
#include <hip/hip_runtime.h>
#include <hip/hip_bf16.h>
#include <math.h>

typedef __bf16 bf16;
typedef __attribute__((ext_vector_type(16))) __bf16 v16bf;
typedef __attribute__((ext_vector_type(8)))  __bf16 v8bf;
typedef __attribute__((ext_vector_type(8)))  float  v8f;

// ---------------------------------------------------------------------------
// helpers
// ---------------------------------------------------------------------------
__device__ inline v8f wmma_bf16f32(v16bf a, v16bf b, v8f c) {
  // D(f32 16x16) = A(bf16 16x32) * B(bf16 32x16) + C
  return __builtin_amdgcn_wmma_f32_16x16x32_bf16(false, a, false, b, (short)0, c,
                                                 false, false);
}

__device__ inline v8f zero8() {
  v8f z;
#pragma unroll
  for (int i = 0; i < 8; i++) z[i] = 0.0f;
  return z;
}

// Build a v16bf A-operand from two contiguous 8-element chunks.
// CDNA5 16-bit A layout (16x32): lane<16 holds K={0..7,16..23},
// lane>=16 holds K={8..15,24..31}; callers pass lo=&row[16*hi], hi=lo+8 (contiguous K=16*hi+e, as B).
__device__ inline v16bf joinA(const bf16* lo, const bf16* hip) {
  v8bf a = *(const v8bf*)lo;
  v8bf b = *(const v8bf*)hip;
  v16bf r;
#pragma unroll
  for (int i = 0; i < 8; i++) { r[i] = a[i]; r[i + 8] = b[i]; }
  return r;
}

// Async global->LDS copy, 16B per lane (GLOBAL_LOAD_ASYNC_TO_LDS_B128).
// Tracked with ASYNCcnt; flat LDS aperture is 4GB-aligned so the low 32 bits
// of a generic pointer to __shared__ are the LDS byte offset.
__device__ inline void async_g2l_b128(void* lds, const void* gsrc) {
  unsigned l = (unsigned)(size_t)lds;
  unsigned long long g = (unsigned long long)(size_t)gsrc;
  asm volatile("global_load_async_to_lds_b128 %0, %1, off"
               :: "v"(l), "v"(g) : "memory");
}
__device__ inline void wait_async0() {
  asm volatile("s_wait_asynccnt 0x0" ::: "memory");
}
// Allow 3 in-flight async ops (the prefetched next tile: 2xA + 1xB).
__device__ inline void wait_async3() {
  asm volatile("s_wait_asynccnt 0x3" ::: "memory");
}
// Intra-wave LDS RAW fence (wave32: LDS ops from one wave are in-order,
// only the counter wait is needed; avoids a block barrier).
__device__ inline void wait_ds0() {
  asm volatile("s_wait_dscnt 0x0" ::: "memory");
}

// ---------------------------------------------------------------------------
// RoPE table: tab[n*16+i] = {cos, sin}(n * 10000^(-i/16)), n<256, i<16
// ---------------------------------------------------------------------------
__global__ void k_rope_tab(float2* __restrict__ tab) {
  int t = blockIdx.x * 256 + threadIdx.x;  // 4096 entries
  int n = t >> 4, i = t & 15;
  float ang = (float)n * powf(10000.0f, -(float)i * (1.0f / 16.0f));
  tab[t] = make_float2(cosf(ang), sinf(ang));
}

// ---------------------------------------------------------------------------
// Weight prep: W f32 (256 x NC, row-major) -> Wg bf16 (NC x 256, row-major)
// i.e. transpose + convert, so the GEMM B tile is a straight async copy.
// grid (NC/32, 8), block (32,8)
// ---------------------------------------------------------------------------
__global__ void k_wt(const float* __restrict__ W, bf16* __restrict__ Wg, int NC) {
  __shared__ float tile[32][33];
  int c0 = blockIdx.x * 32, k0 = blockIdx.y * 32;
  int tx = threadIdx.x, ty = threadIdx.y;
#pragma unroll
  for (int r = 0; r < 32; r += 8)
    tile[tx][ty + r] = W[(size_t)(k0 + ty + r) * NC + c0 + tx];
  __syncthreads();
#pragma unroll
  for (int r = 0; r < 32; r += 8)
    Wg[(size_t)(c0 + ty + r) * 256 + k0 + tx] = (bf16)tile[ty + r][tx];
}

// ---------------------------------------------------------------------------
// x (B,D,T,F) f32 -> actA bf16 as tokens ((b*F+f)*T+t, d)   [stage-1 layout]
// ---------------------------------------------------------------------------
__global__ void k_tin(const float* __restrict__ x, bf16* __restrict__ out) {
  __shared__ float tile[32][33];
  int bt = blockIdx.x;
  int b = bt >> 8, t = bt & 255;
  int f0 = blockIdx.y * 32, d0 = blockIdx.z * 32;
  int tx = threadIdx.x, ty = threadIdx.y;
#pragma unroll
  for (int r = 0; r < 32; r += 8) {
    int d = d0 + ty + r;
    tile[tx][ty + r] = x[(((size_t)b * 256 + d) * 256 + t) * 64 + f0 + tx];
  }
  __syncthreads();
#pragma unroll
  for (int r = 0; r < 32; r += 8) {
    int f = f0 + ty + r;
    out[(((size_t)b * 64 + f) * 256 + t) * 256 + d0 + tx] = (bf16)tile[ty + r][tx];
  }
}

// ---------------------------------------------------------------------------
// Cp f32 tokens ((b*T+t)*F+f, d) -> d_out f32 (B,D,T,F)
// ---------------------------------------------------------------------------
__global__ void k_tout(const float* __restrict__ Cp, float* __restrict__ out) {
  __shared__ float tile[32][33];
  int bt = blockIdx.x;
  int b = bt >> 8, t = bt & 255;
  int f0 = blockIdx.y * 32, d0 = blockIdx.z * 32;
  int tx = threadIdx.x, ty = threadIdx.y;
#pragma unroll
  for (int r = 0; r < 32; r += 8) {
    int f = f0 + ty + r;
    tile[ty + r][tx] = Cp[(((size_t)b * 256 + t) * 64 + f) * 256 + d0 + tx];
  }
  __syncthreads();
#pragma unroll
  for (int r = 0; r < 32; r += 8) {
    int d = d0 + ty + r;
    out[(((size_t)b * 256 + d) * 256 + t) * 64 + f0 + tx] = tile[tx][ty + r];
  }
}

// ---------------------------------------------------------------------------
// Generic bf16 GEMM: C(Mtot x NC) = A(Mtot x 256) * Wg^T, fp32 accum.
// Wg is pre-transposed (NC x 256).  Block tile 128x64, K-step 32, 8 waves
// each computing 2x2 16x16 WMMA tiles.  LDS tiles are double-buffered and
// staged with GLOBAL_LOAD_ASYNC_TO_LDS_B128; the next tile's DMA overlaps
// the current tile's WMMAs (partial s_wait_asynccnt).
// mode 0: QKV epilogue (NC=768): split q/k/v, table-driven RoPE on q/k;
//         q,k -> (seq,head,n,32); v -> transposed (seq,head,hd,n).
// mode 1: plain fp32 store (projection, NC=256).
// ---------------------------------------------------------------------------
__global__ __launch_bounds__(256) void k_gemm(
    const bf16* __restrict__ A, const bf16* __restrict__ Wg,
    const float2* __restrict__ ropetab, int NC, int mode, int lshift,
    bf16* __restrict__ Qb, bf16* __restrict__ Kb, bf16* __restrict__ Vb,
    float* __restrict__ Cout) {
  __shared__ __align__(32) bf16 As[2][128 * 40];  // 80B rows
  __shared__ __align__(32) bf16 Bs[2][64 * 48];   // 96B rows
  int tid = threadIdx.x, lane = tid & 31, wave = tid >> 5;
  int nl = lane & 15, hi = lane >> 4;
  int waveM = wave >> 1, waveN = wave & 1;
  int bm = blockIdx.x, bn = blockIdx.y;

  v8f acc[2][2];
#pragma unroll
  for (int i = 0; i < 2; i++)
#pragma unroll
    for (int j = 0; j < 2; j++) acc[i][j] = zero8();

  int rA = tid >> 1, hh = tid & 1;          // A tile: 2 threads/row
  int cB = tid >> 2, pB = tid & 3;          // B tile: 4 chunks/row
  const bf16* srcA0 = A + (size_t)(bm * 128 + rA) * 256 + hh * 16;
  const bf16* srcB0 = Wg + (size_t)(bn * 64 + cB) * 256 + pB * 8;

  auto issue = [&](int kb, int buf) {
    bf16* dA = &As[buf][rA * 40 + hh * 16];
    async_g2l_b128(dA, srcA0 + kb);
    async_g2l_b128(dA + 8, srcA0 + kb + 8);
    async_g2l_b128(&Bs[buf][cB * 48 + pB * 8], srcB0 + kb);
  };

  issue(0, 0);
  for (int it = 0; it < 8; it++) {
    int buf = it & 1;
    if (it < 7) {
      issue((it + 1) * 32, buf ^ 1);  // prefetch next tile into other buffer
      wait_async3();                  // current tile landed; next stays in flight
    } else {
      wait_async0();
    }
    __syncthreads();

    v16bf afr[2], bfr[2];
#pragma unroll
    for (int tm = 0; tm < 2; tm++) {
      const bf16* p = &As[buf][(waveM * 32 + tm * 16 + nl) * 40 + 16 * hi];
      afr[tm] = joinA(p, p + 8);
    }
#pragma unroll
    for (int tn = 0; tn < 2; tn++) {
      // B layout (32x16): lane holds column nl, K = e + 16*hi
      bfr[tn] = *(const v16bf*)&Bs[buf][(waveN * 32 + tn * 16 + nl) * 48 + 16 * hi];
    }
#pragma unroll
    for (int tm = 0; tm < 2; tm++)
#pragma unroll
      for (int tn = 0; tn < 2; tn++)
        acc[tm][tn] = wmma_bf16f32(afr[tm], bfr[tn], acc[tm][tn]);
    __syncthreads();  // all waves done reading buf -> safe to overwrite
  }

  int Lm = (1 << lshift) - 1;
#pragma unroll
  for (int tm = 0; tm < 2; tm++) {
#pragma unroll
    for (int tn = 0; tn < 2; tn++) {
      int col = bn * 64 + waveN * 32 + tn * 16 + nl;
#pragma unroll
      for (int r = 0; r < 8; r++) {
        int row = bm * 128 + waveM * 32 + tm * 16 + r + 8 * hi;
        float v = acc[tm][tn][r];
        if (mode == 1) {
          Cout[(size_t)row * NC + col] = v;
        } else {
          int part = col >> 8;  // 0=q 1=k 2=v
          int w = col & 255;
          int head = w >> 5, hd = w & 31;
          int seq = row >> lshift, n = row & Lm;
          if (part == 2) {  // V stored transposed: (seq,head,hd,n)
            size_t dstV = (((size_t)(seq * 8 + head) * 32 + hd) << lshift) + n;
            Vb[dstV] = (bf16)v;
          } else {
            size_t dst = (((size_t)(seq * 8 + head) << lshift) + n) * 32 + hd;
            float2 cs = ropetab[(n << 4) + (hd >> 1)];
            float other = __shfl_xor(v, 1, 32);  // RoPE pair partner
            float ov = (hd & 1) ? (v * cs.x + other * cs.y)
                                : (v * cs.x - other * cs.y);
            if (part == 0) Qb[dst] = (bf16)ov; else Kb[dst] = (bf16)ov;
          }
        }
      }
    }
  }
}

// ---------------------------------------------------------------------------
// Attention: one workgroup per (seq, head).
// Q,K: (seq,head,L,32) bf16; V: (seq,head,32,L) bf16 (pre-transposed).
// K and V^T async-staged into LDS; each wave owns 16-row Q strips, streams
// K strips in pairs (K-dim 32 per PV WMMA) with fp32 online softmax.  P's
// C->A re-layout bounces through a per-wave LDS tile guarded by s_wait_dscnt.
// The output store applies the inter-stage token permutation (pmode).
// ---------------------------------------------------------------------------
__global__ void k_attn(const bf16* __restrict__ Q, const bf16* __restrict__ K,
                       const bf16* __restrict__ V, bf16* __restrict__ O, int L,
                       float scale, int pmode) {
  __shared__ __align__(32) bf16 Ks[256 * 48];     // K rows, 96B stride
  __shared__ __align__(32) bf16 Vt[32 * 272];     // V^T rows, 544B stride
  __shared__ __align__(32) bf16 Pscr[8][16 * 32]; // per-wave P scratch
  int seq = blockIdx.x >> 3, head = blockIdx.x & 7;
  size_t base = (size_t)(seq * 8 + head) * L * 32;
  const bf16* Qp = Q + base;
  const bf16* Kp = K + base;
  const bf16* Vp = V + base;
  int tid = threadIdx.x, nthr = blockDim.x;

  int lc = (L == 256) ? 5 : 3;  // log2(L/8): 16B chunks per V^T row
  for (int i = tid; i < L * 4; i += nthr) {  // K rows: 4 chunks/row
    int r = i >> 2, p = i & 3;
    async_g2l_b128(&Ks[r * 48 + p * 8], Kp + i * 8);
  }
  for (int i = tid; i < L * 4; i += nthr) {  // V^T rows
    int c = i >> lc, p = i & ((1 << lc) - 1);
    async_g2l_b128(&Vt[c * 272 + p * 8], Vp + i * 8);
  }
  wait_async0();
  __syncthreads();

  int lane = tid & 31, wave = tid >> 5, nl = lane & 15, hi = lane >> 4;
  int nwaves = nthr >> 5, nstrips = L >> 4;

  for (int strip = wave; strip < nstrips; strip += nwaves) {
    const bf16* qrow = Qp + (size_t)(strip * 16 + nl) * 32 + 16 * hi;
    v16bf aq = joinA(qrow, qrow + 8);
    float m[8], l[8];
    v8f o0 = zero8(), o1 = zero8();
#pragma unroll
    for (int r = 0; r < 8; r++) { m[r] = -3.0e38f; l[r] = 0.0f; }

    for (int j = 0; j < nstrips; j += 2) {
      v16bf b0 = *(const v16bf*)&Ks[(j * 16 + nl) * 48 + 16 * hi];
      v16bf b1 = *(const v16bf*)&Ks[((j + 1) * 16 + nl) * 48 + 16 * hi];
      v8f s0 = wmma_bf16f32(aq, b0, zero8());
      v8f s1 = wmma_bf16f32(aq, b1, zero8());
      bf16* pw = Pscr[wave];
#pragma unroll
      for (int r = 0; r < 8; r++) {
        float x0 = s0[r] * scale, x1 = s1[r] * scale;
        float t = fmaxf(x0, x1);
        t = fmaxf(t, __shfl_xor(t, 1, 32));
        t = fmaxf(t, __shfl_xor(t, 2, 32));
        t = fmaxf(t, __shfl_xor(t, 4, 32));
        t = fmaxf(t, __shfl_xor(t, 8, 32));  // row max (16-lane half)
        float mn = fmaxf(m[r], t);
        float alpha = __expf(m[r] - mn);
        m[r] = mn;
        float p0 = __expf(x0 - mn), p1 = __expf(x1 - mn);
        float s = p0 + p1;
        s += __shfl_xor(s, 1, 32);
        s += __shfl_xor(s, 2, 32);
        s += __shfl_xor(s, 4, 32);
        s += __shfl_xor(s, 8, 32);           // row sum
        l[r] = l[r] * alpha + s;
        o0[r] *= alpha;
        o1[r] *= alpha;
        int rr = r + 8 * hi;                 // C-layout row of this reg
        pw[rr * 32 + nl] = (bf16)p0;
        pw[rr * 32 + 16 + nl] = (bf16)p1;
      }
      wait_ds0();  // intra-wave LDS RAW: stores -> transposed reads
      const bf16* pr = &pw[nl * 32 + 16 * hi];
      v16bf ap = joinA(pr, pr + 8);          // P as A operand (16x32)
      v16bf bv0 = *(const v16bf*)&Vt[nl * 272 + j * 16 + 16 * hi];
      v16bf bv1 = *(const v16bf*)&Vt[(16 + nl) * 272 + j * 16 + 16 * hi];
      o0 = wmma_bf16f32(ap, bv0, o0);
      o1 = wmma_bf16f32(ap, bv1, o1);
    }
#pragma unroll
    for (int r = 0; r < 8; r++) {
      int row = strip * 16 + r + 8 * hi;
      float inv = 1.0f / l[r];
      unsigned tok;
      if (pmode == 0) {        // stage1 ((b f) t) -> stage2 ((b t) f)
        unsigned b = seq >> 6, f = seq & 63;
        tok = (b * 256u + row) * 64u + f;
      } else if (pmode == 1) { // stage2 ((b t) f) -> stage3 windows
        unsigned b = seq >> 8, t = seq & 255;
        unsigned ti = t >> 3, t2 = t & 7, fi = (unsigned)row >> 3, f2 = row & 7;
        tok = ((b * 32u + ti) * 8u + fi) * 64u + t2 * 8u + f2;
      } else {                 // stage3 windows -> proj ((b t) f)
        unsigned b = seq >> 8, ti = (seq >> 3) & 31, fi = seq & 7;
        unsigned t2 = (unsigned)row >> 3, f2 = row & 7;
        tok = (b * 256u + ti * 8u + t2) * 64u + fi * 8u + f2;
      }
      size_t ob = (size_t)tok * 256 + head * 32;
      O[ob + nl] = (bf16)(o0[r] * inv);
      O[ob + 16 + nl] = (bf16)(o1[r] * inv);
    }
  }
}

// ---------------------------------------------------------------------------
// launch
// ---------------------------------------------------------------------------
extern "C" void kernel_launch(void* const* d_in, const int* in_sizes, int n_in,
                              void* d_out, int out_size, void* d_ws,
                              size_t ws_size, hipStream_t stream) {
  const float* x  = (const float*)d_in[0];
  const float* W1 = (const float*)d_in[1];
  const float* W2 = (const float*)d_in[2];
  const float* W3 = (const float*)d_in[3];
  const float* Wp = (const float*)d_in[4];
  (void)in_sizes; (void)n_in; (void)out_size; (void)ws_size;

  char* ws = (char*)d_ws;
  size_t off = 0;
  auto alloc = [&](size_t bytes) -> char* {
    char* p = ws + off;
    off += (bytes + 255) & ~(size_t)255;
    return p;
  };
  const size_t ND = 65536ull * 256ull;
  float2* rtab = (float2*)alloc(4096 * sizeof(float2));
  bf16* Wg1 = (bf16*)alloc(196608 * 2);
  bf16* Wg2 = (bf16*)alloc(196608 * 2);
  bf16* Wg3 = (bf16*)alloc(196608 * 2);
  bf16* Wgp = (bf16*)alloc(65536 * 2);
  bf16* actA = (bf16*)alloc(ND * 2);
  bf16* Qb = (bf16*)alloc(ND * 2);
  bf16* Kb = (bf16*)alloc(ND * 2);   // contiguous after Qb
  bf16* Vb = (bf16*)alloc(ND * 2);
  float* Cp = (float*)Qb;            // proj output aliases Qb+Kb (64 MB)

  const float scale = 0.17677669529663688f;  // 1/sqrt(32)

  k_rope_tab<<<16, 256, 0, stream>>>(rtab);
  k_wt<<<dim3(24, 8), dim3(32, 8), 0, stream>>>(W1, Wg1, 768);
  k_wt<<<dim3(24, 8), dim3(32, 8), 0, stream>>>(W2, Wg2, 768);
  k_wt<<<dim3(24, 8), dim3(32, 8), 0, stream>>>(W3, Wg3, 768);
  k_wt<<<dim3(8, 8), dim3(32, 8), 0, stream>>>(Wp, Wgp, 256);

  k_tin<<<dim3(1024, 2, 8), dim3(32, 8), 0, stream>>>(x, actA);

  // stage 1: time axis (L=256); attn writes stage-2 layout into actA
  k_gemm<<<dim3(512, 12), 256, 0, stream>>>(actA, Wg1, rtab, 768, 0, 8, Qb, Kb,
                                            Vb, nullptr);
  k_attn<<<2048, 256, 0, stream>>>(Qb, Kb, Vb, actA, 256, scale, 0);

  // stage 2: freq axis (L=64); attn writes stage-3 window layout
  k_gemm<<<dim3(512, 12), 256, 0, stream>>>(actA, Wg2, rtab, 768, 0, 6, Qb, Kb,
                                            Vb, nullptr);
  k_attn<<<8192, 128, 0, stream>>>(Qb, Kb, Vb, actA, 64, scale, 1);

  // stage 3: 8x8 windows (L=64); attn writes proj layout
  k_gemm<<<dim3(512, 12), 256, 0, stream>>>(actA, Wg3, rtab, 768, 0, 6, Qb, Kb,
                                            Vb, nullptr);
  k_attn<<<8192, 128, 0, stream>>>(Qb, Kb, Vb, actA, 64, scale, 2);

  // output projection + final transpose
  k_gemm<<<dim3(512, 4), 256, 0, stream>>>(actA, Wgp, rtab, 256, 1, 0, nullptr,
                                           nullptr, nullptr, Cp);
  k_tout<<<dim3(1024, 2, 8), dim3(32, 8), 0, stream>>>(Cp, (float*)d_out);
}
